// GIN_13426067767700
// MI455X (gfx1250) — matrix operand order repaired and checked by reading, not compiled, not run on hardware
//
#include <hip/hip_runtime.h>
#include <hip/hip_bf16.h>
#include <math.h>

typedef __attribute__((ext_vector_type(16))) __bf16 v16bf;
typedef __attribute__((ext_vector_type(8)))  float  v8f;

#define BN_EPS 1e-5f

// Pack two f32 into one dword holding {bf16(a), bf16(b)} -> exactly one
// v_cvt_pk_bf16_f32. Inline asm keeps the SLP vectorizer from re-pairing
// operands and emitting v_perm_b32 fixups.
static __device__ __forceinline__ unsigned pk_bf16(float a, float b) {
    unsigned r;
    asm("v_cvt_pk_bf16_f32 %0, %1, %2" : "=v"(r) : "v"(a), "v"(b));
    return r;
}

// ---------------------------------------------------------------------------
// BatchNorm statistics: per-channel sum and sum-of-squares of X*importance.
// blockDim.x == 128 (one thread per channel), grid-stride over rows.
// ---------------------------------------------------------------------------
__global__ __launch_bounds__(128)
void bn_stats_kernel(const float* __restrict__ X, const float* __restrict__ imp,
                     int n, float* __restrict__ stats /* [256] */) {
    const int c = threadIdx.x;           // channel 0..127
    float s = 0.f, s2 = 0.f;
    for (int r = blockIdx.x; r < n; r += gridDim.x) {
        float v = X[(size_t)r * 128 + c] * imp[r];
        s  += v;
        s2 += v * v;
    }
    unsafeAtomicAdd(&stats[c], s);
    unsafeAtomicAdd(&stats[128 + c], s2);
}

// ---------------------------------------------------------------------------
// Finalize: scale = gamma * rsqrt(var+eps); shift = beta - mu*scale.
// ---------------------------------------------------------------------------
__global__ __launch_bounds__(128)
void bn_finalize_kernel(const float* __restrict__ stats,
                        const float* __restrict__ gamma,
                        const float* __restrict__ beta,
                        float* __restrict__ ss /* [256] scale|shift */, float inv_n) {
    const int c = threadIdx.x;
    float mu  = stats[c] * inv_n;
    float var = stats[128 + c] * inv_n - mu * mu;
    float sc  = gamma[c] * rsqrtf(var + BN_EPS);
    ss[c]        = sc;
    ss[128 + c]  = beta[c] - mu * sc;
}

// ---------------------------------------------------------------------------
// Apply BN: h0[r][c] = (X[r][c]*imp[r]) * scale[c] + shift[c]
// ---------------------------------------------------------------------------
__global__ __launch_bounds__(256)
void bn_apply_kernel(const float* __restrict__ X, const float* __restrict__ imp,
                     const float* __restrict__ ss, float* __restrict__ h0,
                     long long total) {
    long long i = (long long)blockIdx.x * blockDim.x + threadIdx.x;
    if (i >= total) return;
    int c = (int)(i & 127);
    long long r = i >> 7;
    float v = X[i] * imp[r];
    h0[i] = v * ss[c] + ss[128 + c];
}

// ---------------------------------------------------------------------------
// Edge scatter: agg[dst[e]][c..c+3] += H[src[e]][c..c+3]   (float4 chunks)
// One thread = one (edge, 4-channel group). Consecutive threads cover an
// edge's channels contiguously -> coalesced float4 gathers.
// ---------------------------------------------------------------------------
template <int K>
__global__ __launch_bounds__(256)
void gin_scatter_kernel(const float* __restrict__ H, int h_stride,
                        const int* __restrict__ src, const int* __restrict__ dst,
                        float* __restrict__ agg, long long n_work) {
    long long tid = (long long)blockIdx.x * blockDim.x + threadIdx.x;
    if (tid >= n_work) return;
    const int GRP = K / 4;
    int g = (int)(tid % GRP);
    long long e = tid / GRP;
    int s = src[e];
    int d = dst[e];
    const float4 v = *(const float4*)(H + (size_t)s * h_stride + g * 4);
    float* p = agg + (size_t)d * K + g * 4;
    unsafeAtomicAdd(p + 0, v.x);
    unsafeAtomicAdd(p + 1, v.y);
    unsafeAtomicAdd(p + 2, v.z);
    unsafeAtomicAdd(p + 3, v.w);
}

// ---------------------------------------------------------------------------
// Fused GIN layer: out[r][n] = tanh( sum_k (H[r][k] + Agg[r][k]) * W[k][n] + b[n] )
// One wave -> 16x64 output tile via v_wmma_f32_16x16x32_bf16.
// W is staged in LDS PRE-SWIZZLED into WMMA B-fragment order so each lane's
// 16-element fragment is one aligned 32-byte LDS vector load (2x ds_load_b128).
//
// B fragment layout (chunk c, tile t, lane L):
//   col  n = t*16 + (L & 15)
//   k(j)   = c*32 + (L >> 4)*16 + j,   j = 0..15
//   Wswz[ ((c*4 + t)*32 + L)*16 + j ] = bf16( W[k(j)*64 + n] )
// ---------------------------------------------------------------------------
template <int K, bool HAS_AGG, bool HAS_BIAS>
__global__ __launch_bounds__(256)
void gin_gemm_tanh_kernel(const float* __restrict__ H, int h_stride,
                          const float* __restrict__ Agg,
                          const float* __restrict__ W,     // [K][64] row-major
                          const float* __restrict__ bias,  // [64] or null
                          float* __restrict__ out, int out_stride, int n_rows) {
    constexpr int NC = K / 32;                       // K-chunks of 32
    __shared__ __align__(32) __bf16 Wswz[NC * 4 * 32 * 16];
    __shared__ float bl[64];

    // Cooperative fill: one packed bf16 pair (b32 store) per iteration.
    for (int i = threadIdx.x; i < NC * 4 * 32 * 8; i += blockDim.x) {
        const int jp   = i & 7;            // element pair -> j = 2*jp
        const int lane = (i >> 3) & 31;
        const int t    = (i >> 8) & 3;
        const int c    = i >> 10;
        const int nn   = t * 16 + (lane & 15);
        const int k    = c * 32 + (lane >> 4) * 16 + jp * 2;
        ((unsigned*)Wswz)[i] =
            pk_bf16(W[(size_t)k * 64 + nn], W[(size_t)(k + 1) * 64 + nn]);
    }
    if (threadIdx.x < 64)
        bl[threadIdx.x] = HAS_BIAS ? bias[threadIdx.x] : 0.f;
    __syncthreads();

    const int lane = threadIdx.x & 31;
    const int wave = threadIdx.x >> 5;
    const int half = lane >> 4;       // 0: lanes 0-15, 1: lanes 16-31
    const int lrow = lane & 15;

    const int r0 = blockIdx.x * 128 + wave * 16;

    // ---- A fragments: rows r0..r0+15, K split into 32-wide chunks ----
    int arow = r0 + lrow;
    if (arow >= n_rows) arow = n_rows - 1;          // clamp (stores guarded)
    const float* hrow = H + (size_t)arow * h_stride;
    const float* grow = HAS_AGG ? (Agg + (size_t)arow * K) : nullptr;

    union AFrag { v16bf v; unsigned u[8]; };
    AFrag afrag[NC];
#pragma unroll
    for (int c = 0; c < NC; ++c) {
        const int koff = c * 32 + half * 16;
        float4 a0 = *(const float4*)(hrow + koff);
        float4 a1 = *(const float4*)(hrow + koff + 4);
        float4 a2 = *(const float4*)(hrow + koff + 8);
        float4 a3 = *(const float4*)(hrow + koff + 12);
        if (HAS_AGG) {
            float4 g0 = *(const float4*)(grow + koff);
            float4 g1 = *(const float4*)(grow + koff + 4);
            float4 g2 = *(const float4*)(grow + koff + 8);
            float4 g3 = *(const float4*)(grow + koff + 12);
            a0.x += g0.x; a0.y += g0.y; a0.z += g0.z; a0.w += g0.w;
            a1.x += g1.x; a1.y += g1.y; a1.z += g1.z; a1.w += g1.w;
            a2.x += g2.x; a2.y += g2.y; a2.z += g2.z; a2.w += g2.w;
            a3.x += g3.x; a3.y += g3.y; a3.z += g3.z; a3.w += g3.w;
        }
        // elements 0..15 = K koff..koff+15
        afrag[c].u[0] = pk_bf16(a0.x, a0.y);
        afrag[c].u[1] = pk_bf16(a0.z, a0.w);
        afrag[c].u[2] = pk_bf16(a1.x, a1.y);
        afrag[c].u[3] = pk_bf16(a1.z, a1.w);
        afrag[c].u[4] = pk_bf16(a2.x, a2.y);
        afrag[c].u[5] = pk_bf16(a2.z, a2.w);
        afrag[c].u[6] = pk_bf16(a3.x, a3.y);
        afrag[c].u[7] = pk_bf16(a3.z, a3.w);
    }

    const v16bf* Wfrag = (const v16bf*)Wswz;

    // ---- 4 N-tiles of 16 columns each ----
#pragma unroll
    for (int t = 0; t < 4; ++t) {
        const int n = t * 16 + lrow;        // output column for this lane
        v8f acc = {};
#pragma unroll
        for (int c = 0; c < NC; ++c) {
            v16bf bfrag = Wfrag[(c * 4 + t) * 32 + lane];   // 32B aligned LDS load
            acc = __builtin_amdgcn_wmma_f32_16x16x32_bf16(
                false, afrag[c].v, false, bfrag, (short)0, acc, false, false);
        }
        const float bn = bl[n];
#pragma unroll
        for (int r = 0; r < 8; ++r) {
            const int row = r0 + r + half * 8;   // C/D layout: vgpr r -> M=r (+8 for hi lanes)
            if (row < n_rows)
                out[(size_t)row * out_stride + n] = tanhf(acc[r] + bn);
        }
    }
}

// ---------------------------------------------------------------------------
// Host launcher
// ---------------------------------------------------------------------------
extern "C" void kernel_launch(void* const* d_in, const int* in_sizes, int n_in,
                              void* d_out, int out_size, void* d_ws, size_t ws_size,
                              hipStream_t stream) {
    const float* X     = (const float*)d_in[0];
    const float* imp   = (const float*)d_in[1];
    const int*   eidx  = (const int*)d_in[2];
    const float* gamma = (const float*)d_in[3];
    const float* beta  = (const float*)d_in[4];
    const float* W1 = (const float*)d_in[5];  const float* b1 = (const float*)d_in[6];
    const float* W2 = (const float*)d_in[7];  const float* b2 = (const float*)d_in[8];
    const float* W3 = (const float*)d_in[9];  const float* b3 = (const float*)d_in[10];
    const float* W4 = (const float*)d_in[11]; const float* b4 = (const float*)d_in[12];
    const float* W5 = (const float*)d_in[13]; const float* b5 = (const float*)d_in[14];
    const float* Wf = (const float*)d_in[15];

    const int n = in_sizes[0] / 128;              // 100000
    const long long E = in_sizes[2] / 2;          // 1600000
    const int* src = eidx;
    const int* dst = eidx + E;

    // workspace layout
    char*  ws    = (char*)d_ws;
    float* stats = (float*)ws;                    // 256 floats (sum | sumsq)
    float* ss    = stats + 256;                   // 256 floats (scale | shift)
    float* h0    = (float*)(ws + 4096);           // [n][128]
    float* agg   = h0 + (size_t)n * 128;          // [n][128] (reused at width 64)
    float* out   = (float*)d_out;                 // [n][384]

    // ---- BatchNorm (batch statistics) ----
    hipMemsetAsync(stats, 0, 512 * sizeof(float), stream);
    bn_stats_kernel<<<1024, 128, 0, stream>>>(X, imp, n, stats);
    bn_finalize_kernel<<<1, 128, 0, stream>>>(stats, gamma, beta, ss, 1.0f / (float)n);
    {
        long long tot = (long long)n * 128;
        bn_apply_kernel<<<(unsigned)((tot + 255) / 256), 256, 0, stream>>>(X, imp, ss, h0, tot);
    }

    const unsigned gblocks = (unsigned)((n + 127) / 128);

    // ---- Layer 1: K=128, gather from h0 ----
    hipMemsetAsync(agg, 0, (size_t)n * 128 * sizeof(float), stream);
    {
        long long w = E * 32;                     // E * (128/4)
        gin_scatter_kernel<128><<<(unsigned)((w + 255) / 256), 256, 0, stream>>>(
            h0, 128, src, dst, agg, w);
    }
    gin_gemm_tanh_kernel<128, true, true><<<gblocks, 256, 0, stream>>>(
        h0, 128, agg, W1, b1, out + 0, 384, n);

    // ---- Layers 2..5: K=64, gather from previous output columns ----
    const float* Ws[4] = {W2, W3, W4, W5};
    const float* bs[4] = {b2, b3, b4, b5};
    for (int l = 0; l < 4; ++l) {
        const float* Hl = out + l * 64;           // stride 384
        hipMemsetAsync(agg, 0, (size_t)n * 64 * sizeof(float), stream);
        long long w = E * 16;                     // E * (64/4)
        gin_scatter_kernel<64><<<(unsigned)((w + 255) / 256), 256, 0, stream>>>(
            Hl, 384, src, dst, agg, w);
        gin_gemm_tanh_kernel<64, true, true><<<gblocks, 256, 0, stream>>>(
            Hl, 384, agg, Ws[l], bs[l], out + (l + 1) * 64, 384, n);
    }

    // ---- Final: tanh(h5 @ Wf), no aggregation, no bias ----
    gin_gemm_tanh_kernel<64, false, false><<<gblocks, 256, 0, stream>>>(
        out + 4 * 64, 384, nullptr, Wf, nullptr, out + 5 * 64, 384, n);
}
